// VectorQuantizer_16630113370446
// MI455X (gfx1250) — matrix operand order repaired and checked by reading, not compiled, not run on hardware
//
#include <hip/hip_runtime.h>

typedef __attribute__((ext_vector_type(16))) __bf16 v16bf;
typedef __attribute__((ext_vector_type(8)))  __bf16 v8bf;
typedef __attribute__((ext_vector_type(8)))  float  v8f;

#define NUM_CODES 1024
#define CODE_DIM  256
#define N_ROWS    65536          // 64*32*32
#define ROWS_PER_WG 32
#define N_WG      (N_ROWS / ROWS_PER_WG)   // 2048
#define ZPITCH    264            // bf16 row pitch (bank-rotating pad)
#define ZFPITCH   260            // f32 row pitch

// ---- RNE float -> bf16 split helpers (pure bit ops; no reliance on native cvt) ----
__device__ __forceinline__ __bf16 f2bf(float f) {
    unsigned u = __builtin_bit_cast(unsigned, f);
    unsigned r = (u + 0x7FFFu + ((u >> 16) & 1u)) >> 16;
    unsigned short s = (unsigned short)r;
    return __builtin_bit_cast(__bf16, s);
}
__device__ __forceinline__ float bf2f(__bf16 b) {
    unsigned short s = __builtin_bit_cast(unsigned short, b);
    unsigned u = ((unsigned)s) << 16;
    return __builtin_bit_cast(float, u);
}

// =====================================================================
// Kernel 1: codebook prep — bf16 hi/lo planes + ||c||^2 (f32)
// grid: 1024 blocks (one per code) x 256 threads (one per dim)
// =====================================================================
__global__ __launch_bounds__(256) void vq_prep(const float* __restrict__ cb,
                                               __bf16* __restrict__ cb_hi,
                                               __bf16* __restrict__ cb_lo,
                                               float* __restrict__ csq) {
    __shared__ float wsum[8];
    int b = blockIdx.x, t = threadIdx.x;
    float f = cb[b * CODE_DIM + t];
    __bf16 h = f2bf(f);
    cb_hi[b * CODE_DIM + t] = h;
    cb_lo[b * CODE_DIM + t] = f2bf(f - bf2f(h));
    float s = f * f;
    #pragma unroll
    for (int off = 16; off > 0; off >>= 1) s += __shfl_xor(s, off, 32);
    int lane = t & 31, wave = t >> 5;
    if (lane == 0) wsum[wave] = s;
    __syncthreads();
    if (t == 0) {
        float tot = 0.f;
        #pragma unroll
        for (int w = 0; w < 8; w++) tot += wsum[w];
        csq[b] = tot;
    }
}

// =====================================================================
// Kernel 2: main — WMMA distance GEMM + argmin + gather + loss partials
// grid: 2048 blocks x 256 threads; block owns 32 rows of z_flat
// =====================================================================
__global__ __launch_bounds__(256) void vq_main(const float*  __restrict__ z,
                                               const float*  __restrict__ cb,
                                               const __bf16* __restrict__ cb_hi,
                                               const __bf16* __restrict__ cb_lo,
                                               const float*  __restrict__ csq,
                                               float* __restrict__ out_zq,
                                               float* __restrict__ out_idx,
                                               float* __restrict__ partials) {
    __shared__ __bf16 zhi[ROWS_PER_WG * ZPITCH];
    __shared__ __bf16 zlo[ROWS_PER_WG * ZPITCH];
    __shared__ float  zf[ROWS_PER_WG * ZFPITCH];
    __shared__ float  wmin[8][ROWS_PER_WG];
    __shared__ int    widx[8][ROWS_PER_WG];
    __shared__ int    fidx[ROWS_PER_WG];
    __shared__ float  wsum[8];

    const int t = threadIdx.x;
    const int wg = blockIdx.x;
    const int lane = t & 31, wave = t >> 5;
    const long base_row = (long)wg * ROWS_PER_WG;

    // ---- stage z tile: f32 + bf16 hi/lo split into LDS ----
    {
        int r  = t >> 3;                 // 0..31
        int d0 = (t & 7) << 5;           // 0,32,...,224
        const float* src = z + (base_row + r) * CODE_DIM + d0;
        #pragma unroll
        for (int i = 0; i < 32; i += 4) {
            float4 v = *(const float4*)(src + i);
            float* zr = &zf[r * ZFPITCH + d0 + i];
            zr[0] = v.x; zr[1] = v.y; zr[2] = v.z; zr[3] = v.w;
            int o = r * ZPITCH + d0 + i;
            auto cv = [&](float fv, int oo) {
                __bf16 h = f2bf(fv);
                zhi[oo] = h;
                zlo[oo] = f2bf(fv - bf2f(h));
            };
            cv(v.x, o + 0); cv(v.y, o + 1); cv(v.z, o + 2); cv(v.w, o + 3);
        }
    }
    __syncthreads();

    // ---- WMMA distance phase ----
    float mn0[8], mn1[8]; int mi0[8], mi1[8];
    #pragma unroll
    for (int j = 0; j < 8; j++) { mn0[j] = 3.402823466e38f; mn1[j] = 3.402823466e38f; mi0[j] = 0; mi1[j] = 0; }

    const int lrow    = lane & 15;                 // A-matrix row within 16-row tile
    const int alo_off = (lane < 16) ? 0 : 8;       // K offset for frag elems 0..7
    const int ahi_off = alo_off + 16;              // K offset for frag elems 8..15
    const int bk_off  = (lane < 16) ? 0 : 8;       // B: K offset for frag elems 0..7 (elems 8..15 at +16)
    const int coln    = lane & 15;                 // B column within 16-code tile

    for (int ct = 0; ct < 8; ct++) {               // 8 x 16 codes per wave
        v8f acc0 = {}; v8f acc1 = {};
        const int c = wave * 128 + ct * 16 + coln; // this lane's code index
        const __bf16* bhb = cb_hi + (long)c * CODE_DIM + bk_off;
        const __bf16* blb = cb_lo + (long)c * CODE_DIM + bk_off;
        #pragma unroll
        for (int kt = 0; kt < 8; kt++) {           // K = 256 in steps of 32
            __builtin_prefetch(bhb + (kt + 1) * 32, 0, 0);
            v16bf bh = __builtin_shufflevector(*(const v8bf*)(bhb + kt * 32), *(const v8bf*)(bhb + kt * 32 + 16),
                                               0,1,2,3,4,5,6,7,8,9,10,11,12,13,14,15);
            v16bf bl = __builtin_shufflevector(*(const v8bf*)(blb + kt * 32), *(const v8bf*)(blb + kt * 32 + 16),
                                               0,1,2,3,4,5,6,7,8,9,10,11,12,13,14,15);

            const __bf16* a0h = &zhi[lrow * ZPITCH + kt * 32];
            const __bf16* a0l = &zlo[lrow * ZPITCH + kt * 32];
            const __bf16* a1h = &zhi[(16 + lrow) * ZPITCH + kt * 32];
            const __bf16* a1l = &zlo[(16 + lrow) * ZPITCH + kt * 32];
            v16bf A0h = __builtin_shufflevector(*(const v8bf*)(a0h + alo_off), *(const v8bf*)(a0h + ahi_off),
                                                0,1,2,3,4,5,6,7,8,9,10,11,12,13,14,15);
            v16bf A0l = __builtin_shufflevector(*(const v8bf*)(a0l + alo_off), *(const v8bf*)(a0l + ahi_off),
                                                0,1,2,3,4,5,6,7,8,9,10,11,12,13,14,15);
            v16bf A1h = __builtin_shufflevector(*(const v8bf*)(a1h + alo_off), *(const v8bf*)(a1h + ahi_off),
                                                0,1,2,3,4,5,6,7,8,9,10,11,12,13,14,15);
            v16bf A1l = __builtin_shufflevector(*(const v8bf*)(a1l + alo_off), *(const v8bf*)(a1l + ahi_off),
                                                0,1,2,3,4,5,6,7,8,9,10,11,12,13,14,15);

            // split-precision dot: hi*hi + hi*lo + lo*hi
            acc0 = __builtin_amdgcn_wmma_f32_16x16x32_bf16(false, A0h, false, bh, (short)0, acc0, false, false);
            acc0 = __builtin_amdgcn_wmma_f32_16x16x32_bf16(false, A0h, false, bl, (short)0, acc0, false, false);
            acc0 = __builtin_amdgcn_wmma_f32_16x16x32_bf16(false, A0l, false, bh, (short)0, acc0, false, false);
            acc1 = __builtin_amdgcn_wmma_f32_16x16x32_bf16(false, A1h, false, bh, (short)0, acc1, false, false);
            acc1 = __builtin_amdgcn_wmma_f32_16x16x32_bf16(false, A1h, false, bl, (short)0, acc1, false, false);
            acc1 = __builtin_amdgcn_wmma_f32_16x16x32_bf16(false, A1l, false, bh, (short)0, acc1, false, false);
        }
        // score = ||c||^2 - 2 z.c  (||z||^2 is row-constant -> irrelevant to argmin)
        const float cs = csq[c];
        #pragma unroll
        for (int j = 0; j < 8; j++) {
            float s0 = cs - 2.0f * acc0[j];
            if (s0 < mn0[j]) { mn0[j] = s0; mi0[j] = c; }   // c grows with ct -> first-index ties kept
            float s1 = cs - 2.0f * acc1[j];
            if (s1 < mn1[j]) { mn1[j] = s1; mi1[j] = c; }
        }
    }

    // ---- cross-lane argmin within each 16-lane half (same row group) ----
    #pragma unroll
    for (int off = 1; off < 16; off <<= 1) {
        #pragma unroll
        for (int j = 0; j < 8; j++) {
            float om = __shfl_xor(mn0[j], off, 32); int oi = __shfl_xor(mi0[j], off, 32);
            if (om < mn0[j] || (om == mn0[j] && oi < mi0[j])) { mn0[j] = om; mi0[j] = oi; }
            om = __shfl_xor(mn1[j], off, 32); oi = __shfl_xor(mi1[j], off, 32);
            if (om < mn1[j] || (om == mn1[j] && oi < mi1[j])) { mn1[j] = om; mi1[j] = oi; }
        }
    }
    if ((lane & 15) == 0) {
        int half = (lane >> 4) ? 8 : 0;   // lanes 0..15 hold rows M=j, lanes 16..31 hold M=j+8
        #pragma unroll
        for (int j = 0; j < 8; j++) {
            wmin[wave][half + j]      = mn0[j]; widx[wave][half + j]      = mi0[j];
            wmin[wave][16 + half + j] = mn1[j]; widx[wave][16 + half + j] = mi1[j];
        }
    }
    __syncthreads();

    // ---- argmin across the 8 waves (each covered a disjoint 128-code band) ----
    if (t < ROWS_PER_WG) {
        float m = wmin[0][t]; int ix = widx[0][t];
        #pragma unroll
        for (int w = 1; w < 8; w++) {
            float om = wmin[w][t]; int oi = widx[w][t];
            if (om < m || (om == m && oi < ix)) { m = om; ix = oi; }
        }
        fidx[t] = ix;
        out_idx[base_row + t] = (float)ix;
    }
    __syncthreads();

    // ---- gather z_q (= z_q_st numerically) + SSE partial ----
    float sse = 0.f;
    {
        int r = t >> 3, d0 = (t & 7) << 5;
        int ci = fidx[r];
        const float4* q = (const float4*)(cb + (long)ci * CODE_DIM + d0);
        float4* o = (float4*)(out_zq + (base_row + r) * CODE_DIM + d0);
        const float* zr = &zf[r * ZFPITCH + d0];
        #pragma unroll
        for (int i = 0; i < 8; i++) {
            float4 v = q[i];
            o[i] = v;
            float dx = zr[4*i+0] - v.x, dy = zr[4*i+1] - v.y;
            float dz = zr[4*i+2] - v.z, dw = zr[4*i+3] - v.w;
            sse += dx*dx + dy*dy + dz*dz + dw*dw;
        }
    }
    #pragma unroll
    for (int off = 16; off > 0; off >>= 1) sse += __shfl_xor(sse, off, 32);
    if (lane == 0) wsum[wave] = sse;
    __syncthreads();
    if (t == 0) {
        float s = 0.f;
        #pragma unroll
        for (int w = 0; w < 8; w++) s += wsum[w];
        partials[wg] = s;   // plain store: deterministic (no float atomics)
    }
}

// =====================================================================
// Kernel 3: reduce 2048 partials -> vq_loss = 1.25 * mean(sse)
// =====================================================================
__global__ __launch_bounds__(256) void vq_loss_k(const float* __restrict__ partials,
                                                 float* __restrict__ out_loss) {
    __shared__ float wsum[8];
    int t = threadIdx.x;
    float s = 0.f;
    for (int i = t; i < N_WG; i += 256) s += partials[i];
    #pragma unroll
    for (int off = 16; off > 0; off >>= 1) s += __shfl_xor(s, off, 32);
    if ((t & 31) == 0) wsum[t >> 5] = s;
    __syncthreads();
    if (t == 0) {
        float tot = 0.f;
        #pragma unroll
        for (int w = 0; w < 8; w++) tot += wsum[w];
        *out_loss = tot * (1.25f / ((float)N_ROWS * (float)CODE_DIM));
    }
}

extern "C" void kernel_launch(void* const* d_in, const int* in_sizes, int n_in,
                              void* d_out, int out_size, void* d_ws, size_t ws_size,
                              hipStream_t stream) {
    const float* z  = (const float*)d_in[0];   // [65536, 256]
    const float* cb = (const float*)d_in[1];   // [1024, 256]
    float* out = (float*)d_out;

    // workspace layout: cb_hi (512KB) | cb_lo (512KB) | csq (4KB) | partials (8KB)
    __bf16* cb_hi = (__bf16*)d_ws;
    __bf16* cb_lo = cb_hi + NUM_CODES * CODE_DIM;
    float*  csq   = (float*)(cb_lo + NUM_CODES * CODE_DIM);
    float*  parts = csq + NUM_CODES;

    // d_out layout (reference return order): z_q_st | vq_loss | indices(as float)
    float* out_zq   = out;
    float* out_loss = out + (long)N_ROWS * CODE_DIM;
    float* out_idx  = out_loss + 1;

    vq_prep<<<NUM_CODES, 256, 0, stream>>>(cb, cb_hi, cb_lo, csq);
    vq_main<<<N_WG, 256, 0, stream>>>(z, cb, cb_hi, cb_lo, csq, out_zq, out_idx, parts);
    vq_loss_k<<<1, 256, 0, stream>>>(parts, out_loss);
}
